// Generator_56470230008596
// MI455X (gfx1250) — matrix operand order repaired and checked by reading, not compiled, not run on hardware
//
#include <hip/hip_runtime.h>
#include <hip/hip_bf16.h>
#include <cstdint>
#include <cstddef>

#define BQ 8
#define TQ 128
#define SQ 512
#define DQ 512
#define VQ 32000
#define LDO (2 * VQ) /* 64000 */

typedef __attribute__((ext_vector_type(16))) __bf16 v16bf;
typedef __attribute__((ext_vector_type(8)))  __bf16 v8bf;
typedef __attribute__((ext_vector_type(8)))  float  v8f;
typedef __attribute__((ext_vector_type(4)))  unsigned int u32x4;
typedef __attribute__((ext_vector_type(8)))  int i32x8;
typedef __attribute__((ext_vector_type(4)))  int i32x4;

#if defined(__HIP_DEVICE_COMPILE__) && \
    __has_builtin(__builtin_amdgcn_tensor_load_to_lds) && \
    __has_builtin(__builtin_amdgcn_s_wait_tensorcnt)
#define GEN_USE_TDM 1
#else
#define GEN_USE_TDM 0
#endif

// ---------------- WMMA fragment loaders (bf16, 16x16x32) ----------------
// A (16x32, MxK): lane<16 -> M=lane, K=0..15;
// lane>=16 -> M=lane-16, K=16..31; 2 bf16 packed per VGPR.
__device__ __forceinline__ v16bf load_fragA(const __bf16* base, int ld, int lane) {
  const int half = lane >> 4;
  const int r    = lane & 15;
  const __bf16* p = base + (size_t)r * ld + half * 16;
  v8bf lo = *reinterpret_cast<const v8bf*>(p);        // K = half*16 + 0..7
  v8bf hi = *reinterpret_cast<const v8bf*>(p + 8);    // K = half*16 + 8..15
  v16bf f;
#pragma unroll
  for (int i = 0; i < 8; ++i) { f[i] = lo[i]; f[i + 8] = hi[i]; }
  return f;
}

// B (32x16, KxN): lane<16 -> N=lane, K=0..15; lane>=16 -> N=lane-16, K=16..31.
__device__ __forceinline__ v16bf load_fragB(const __bf16* base, int ld, int lane) {
  const int half = lane >> 4;
  const int n    = lane & 15;
  const __bf16* p = base + (size_t)n * ld + half * 16;
  v8bf lo = *reinterpret_cast<const v8bf*>(p);
  v8bf hi = *reinterpret_cast<const v8bf*>(p + 8);
  v16bf f;
#pragma unroll
  for (int i = 0; i < 8; ++i) { f[i] = lo[i]; f[i + 8] = hi[i]; }
  return f;
}

// Same B fragment sourced from an LDS tile with 40-element (80 B) row pitch.
#define BPITCH 40
__device__ __forceinline__ v16bf load_fragB_lds(const __bf16* buf, int row0, int lane) {
  const int half = lane >> 4;
  const int n    = lane & 15;
  const __bf16* p = buf + (size_t)(row0 + n) * BPITCH + half * 16;
  v8bf lo = *reinterpret_cast<const v8bf*>(p);   // ds_load_b128
  v8bf hi = *reinterpret_cast<const v8bf*>(p + 8);
  v16bf f;
#pragma unroll
  for (int i = 0; i < 8; ++i) { f[i] = lo[i]; f[i + 8] = hi[i]; }
  return f;
}

#if GEN_USE_TDM
// Issue one TDM transfer: 128 rows x 64 B (32 bf16) tile of W -> LDS,
// data_size=8B, tile_dim0=8 units, tile_dim1=128 rows, row stride 128 units
// (= 1 KB), pad 4 DWORDs after every 16 DWORDs -> 80 B LDS pitch.
__device__ __forceinline__ void tdm_issue_b_tile(uint64_t gaddr, unsigned lds_addr) {
  u32x4 g0 = {0u, 0u, 0u, 0u};
  g0[0] = 1u;                                                     // count=1
  g0[1] = lds_addr;                                               // D#.lds_addr
  g0[2] = (unsigned)gaddr;                                        // global_addr[31:0]
  g0[3] = ((unsigned)(gaddr >> 32) & 0x01FFFFFFu) | 0x80000000u;  // addr[56:32] | type=2
  i32x8 g1 = {0, 0, 0, 0, 0, 0, 0, 0};
  g1[0] = (int)((3u << 16) |   // data_size = 8 B
                (1u << 20) |   // pad_enable
                (3u << 22) |   // pad_interval = 16 DWORDs (64 B)
                (3u << 25));   // pad_amount   = 4 DWORDs (16 B)
  g1[1] = (int)(128u << 16);   // tensor_dim0 = 128 units (lo16)
  g1[2] = (int)(32000u << 16); // tensor_dim1 = 32000 rows (lo16)
  g1[3] = (int)(8u << 16);     // tile_dim0 = 8 units (64 B)
  g1[4] = (int)128;            // tile_dim1 = 128 rows, tile_dim2 = 0
  g1[5] = (int)128;            // tensor_dim0_stride = 128 units (1 KB)
  i32x4 z4 = {0, 0, 0, 0};
#if __clang_major__ >= 23
  i32x8 z8 = {0, 0, 0, 0, 0, 0, 0, 0};
  __builtin_amdgcn_tensor_load_to_lds(g0, g1, z4, z4, z8, 0);
#else
  __builtin_amdgcn_tensor_load_to_lds(g0, g1, z4, z4, 0);
#endif
}
#endif

// ---------------- fp32 -> bf16 conversion ----------------
__global__ void f32_to_bf16_kernel(const float* __restrict__ src,
                                   __bf16* __restrict__ dst, int n) {
  int i = blockIdx.x * blockDim.x + threadIdx.x;
  if (i < n) dst[i] = (__bf16)src[i];
}

// ---------------- GEMM 1: gen_scores = s @ Wo^T + Wo_b -> out[:, :V] ----------------
// M = B*T = 1024, N = V = 32000, K = D = 512. 8 waves/block, tile 64(M) x 128(N).
// B (weight) tile double-buffered in LDS via TDM; A frags loaded direct.
__global__ __launch_bounds__(256)
void gen_scores_gemm(const __bf16* __restrict__ A, const __bf16* __restrict__ W,
                     const float* __restrict__ bias, float* __restrict__ C) {
  __shared__ __align__(128) __bf16 bufB[2][128 * BPITCH];  // 2 x 10 KB
  const int tid   = threadIdx.x;
  const int lane  = tid & 31;
  const int wave  = tid >> 5;
  const int mTile = blockIdx.y * 64 + (wave & 3) * 16;
  const int nTile0 = blockIdx.x * 128;             // block's 128 weight rows
  const int nBase  = nTile0 + (wave >> 2) * 64;
  const int rowB   = (wave >> 2) * 64;             // this wave's rows in the LDS tile
  v8f acc[4] = {};
  const __bf16* aRow = A + (size_t)mTile * DQ;

#if GEN_USE_TDM
  const uint64_t wbase = (uint64_t)(uintptr_t)(W + (size_t)nTile0 * DQ);
  if (wave == 0) {
    tdm_issue_b_tile(wbase, (unsigned)(uintptr_t)&bufB[0][0]);
    __builtin_amdgcn_s_wait_tensorcnt(0);
  }
  __syncthreads();
  int cur = 0;
  for (int k = 0; k < DQ; k += 32) {
    if (wave == 0 && (k + 32) < DQ)
      tdm_issue_b_tile(wbase + (uint64_t)(k + 32) * 2,
                       (unsigned)(uintptr_t)&bufB[cur ^ 1][0]);
    v16bf a = load_fragA(aRow + k, DQ, lane);
#pragma unroll
    for (int j = 0; j < 4; ++j) {
      v16bf b = load_fragB_lds(&bufB[cur][0], rowB + j * 16, lane);
      acc[j] = __builtin_amdgcn_wmma_f32_16x16x32_bf16(
          false, a, false, b, (short)0, acc[j], false, false);
    }
    if (wave == 0 && (k + 32) < DQ) __builtin_amdgcn_s_wait_tensorcnt(0);
    __syncthreads();
    cur ^= 1;
  }
#else
  for (int k = 0; k < DQ; k += 32) {
    // cooperative copy: 128 rows x 32 bf16 (4 x 16 B segs per row) into padded tile
    for (int idx = tid; idx < 128 * 4; idx += 256) {
      const int row = idx >> 2, seg = idx & 3;
      *reinterpret_cast<uint4*>(&bufB[0][row * BPITCH + seg * 8]) =
          *reinterpret_cast<const uint4*>(W + (size_t)(nTile0 + row) * DQ + k + seg * 8);
    }
    __syncthreads();
    v16bf a = load_fragA(aRow + k, DQ, lane);
#pragma unroll
    for (int j = 0; j < 4; ++j) {
      v16bf b = load_fragB_lds(&bufB[0][0], rowB + j * 16, lane);
      acc[j] = __builtin_amdgcn_wmma_f32_16x16x32_bf16(
          false, a, false, b, (short)0, acc[j], false, false);
    }
    __syncthreads();
  }
#endif

  const int half = lane >> 4, r = lane & 15;
#pragma unroll
  for (int j = 0; j < 4; ++j) {
    const int n = nBase + j * 16 + r;
    const float bv = bias[n];
#pragma unroll
    for (int i = 0; i < 8; ++i) {
      const int m = mTile + half * 8 + i;
      C[(size_t)m * LDO + n] = acc[j][i] + bv;
    }
  }
}

// ---------------- GEMM 2: copy_feat = tanh(h @ Wc^T + Wc_b) -> bf16 ----------------
// M = B*S = 4096, N = D = 512, K = D = 512.
__global__ __launch_bounds__(256)
void copy_feat_gemm(const __bf16* __restrict__ A, const __bf16* __restrict__ W,
                    const float* __restrict__ bias, __bf16* __restrict__ C) {
  const int lane  = threadIdx.x & 31;
  const int wave  = threadIdx.x >> 5;
  const int mTile = blockIdx.y * 64 + (wave & 3) * 16;
  const int nBase = blockIdx.x * 128 + (wave >> 2) * 64;
  v8f acc[4] = {};
  const __bf16* aRow = A + (size_t)mTile * DQ;
  for (int k = 0; k < DQ; k += 32) {
    v16bf a = load_fragA(aRow + k, DQ, lane);
#pragma unroll
    for (int j = 0; j < 4; ++j) {
      v16bf b = load_fragB(W + (size_t)(nBase + j * 16) * DQ + k, DQ, lane);
      acc[j] = __builtin_amdgcn_wmma_f32_16x16x32_bf16(
          false, a, false, b, (short)0, acc[j], false, false);
    }
  }
  const int half = lane >> 4, r = lane & 15;
#pragma unroll
  for (int j = 0; j < 4; ++j) {
    const int n = nBase + j * 16 + r;
    const float bv = bias[n];
#pragma unroll
    for (int i = 0; i < 8; ++i) {
      const int m = mTile + half * 8 + i;
      C[(size_t)m * DQ + n] = (__bf16)tanhf(acc[j][i] + bv);
    }
  }
}

// ---------------- GEMM 3 (batched): copy_scores[b] = s[b] @ copy_feat[b]^T ----------------
// Per batch: M = T = 128, N = S = 512, K = D = 512. grid.z = B.
__global__ __launch_bounds__(256)
void copy_scores_gemm(const __bf16* __restrict__ Sbf, const __bf16* __restrict__ CF,
                      float* __restrict__ CS) {
  const int lane  = threadIdx.x & 31;
  const int wave  = threadIdx.x >> 5;
  const int mTile = blockIdx.y * 64 + (wave & 3) * 16;
  const int nBase = blockIdx.x * 128 + (wave >> 2) * 64;
  const __bf16* A  = Sbf + (size_t)blockIdx.z * TQ * DQ;
  const __bf16* Bm = CF + (size_t)blockIdx.z * SQ * DQ;
  float* C = CS + (size_t)blockIdx.z * TQ * SQ;
  v8f acc[4] = {};
  const __bf16* aRow = A + (size_t)mTile * DQ;
  for (int k = 0; k < DQ; k += 32) {
    v16bf a = load_fragA(aRow + k, DQ, lane);
#pragma unroll
    for (int j = 0; j < 4; ++j) {
      v16bf b = load_fragB(Bm + (size_t)(nBase + j * 16) * DQ + k, DQ, lane);
      acc[j] = __builtin_amdgcn_wmma_f32_16x16x32_bf16(
          false, a, false, b, (short)0, acc[j], false, false);
    }
  }
  const int half = lane >> 4, r = lane & 15;
#pragma unroll
  for (int j = 0; j < 4; ++j) {
    const int n = nBase + j * 16 + r;
#pragma unroll
    for (int i = 0; i < 8; ++i) {
      const int m = mTile + half * 8 + i;
      C[(size_t)m * SQ + n] = acc[j][i];
    }
  }
}

// ---------------- Fused softmax + vocab scatter (LDS) + log ----------------
// One 512-thread block per (b,t) row; 32000-entry f32 vocab accumulator in
// dynamic LDS (125 KB; CDNA5 WGP has 320 KB), ds_add_f32 atomics for scatter.
__global__ __launch_bounds__(512)
void softmax_scatter_kernel(const float* __restrict__ CS, const int* __restrict__ seq,
                            float* __restrict__ out) {
  extern __shared__ float vocab[];          // VQ floats
  __shared__ float red[512];
  const int row = blockIdx.x;               // b*T + t
  const int b   = row >> 7;                 // T = 128
  float* gen = out + (size_t)row * LDO;     // [0, V): gen scores in-place
  const float* cs = CS + (size_t)row * SQ;  // copy scores
  const int* ids = seq + b * SQ;
  const int tid = threadIdx.x;

  float mx = -3.402823466e+38f;
  for (int i = tid; i < VQ; i += 512) mx = fmaxf(mx, gen[i]);
  mx = fmaxf(mx, cs[tid]);
  red[tid] = mx;
  __syncthreads();
  for (int st = 256; st > 0; st >>= 1) {
    if (tid < st) red[tid] = fmaxf(red[tid], red[tid + st]);
    __syncthreads();
  }
  mx = red[0];
  __syncthreads();

  float sum = 0.0f;
  for (int i = tid; i < VQ; i += 512) sum += __expf(gen[i] - mx);
  sum += __expf(cs[tid] - mx);
  red[tid] = sum;
  __syncthreads();
  for (int st = 256; st > 0; st >>= 1) {
    if (tid < st) red[tid] += red[tid + st];
    __syncthreads();
  }
  const float invZ = 1.0f / red[0];
  __syncthreads();

  for (int i = tid; i < VQ; i += 512) {
    const float p = __expf(gen[i] - mx) * invZ;
    gen[i] = __logf(p + 1e-20f);
  }

  for (int i = tid; i < VQ; i += 512) vocab[i] = 0.0f;
  __syncthreads();
  {
    const float p = __expf(cs[tid] - mx) * invZ;
    atomicAdd(&vocab[ids[tid]], p);
  }
  __syncthreads();

  float* cp = gen + VQ;
  for (int i = tid; i < VQ; i += 512) cp[i] = __logf(vocab[i] + 1e-20f);
}

// ---------------- launch ----------------
extern "C" void kernel_launch(void* const* d_in, const int* in_sizes, int n_in,
                              void* d_out, int out_size, void* d_ws, size_t ws_size,
                              hipStream_t stream) {
  (void)in_sizes; (void)n_in; (void)out_size; (void)ws_size;
  const float* s    = (const float*)d_in[0];
  const float* h    = (const float*)d_in[1];
  const int*   seq  = (const int*)d_in[2];
  const float* Wo_w = (const float*)d_in[3];
  const float* Wo_b = (const float*)d_in[4];
  const float* Wc_w = (const float*)d_in[5];
  const float* Wc_b = (const float*)d_in[6];
  float* out = (float*)d_out;

  char* ws = (char*)d_ws;
  size_t off = 0;
  auto alloc = [&](size_t bytes) -> void* {
    void* p = ws + off;
    off = (off + bytes + 255) & ~(size_t)255;
    return p;
  };
  __bf16* s_bf  = (__bf16*)alloc((size_t)BQ * TQ * DQ * 2);  // 1 MB
  __bf16* h_bf  = (__bf16*)alloc((size_t)BQ * SQ * DQ * 2);  // 4 MB
  __bf16* Wo_bf = (__bf16*)alloc((size_t)VQ * DQ * 2);       // 32.8 MB
  __bf16* Wc_bf = (__bf16*)alloc((size_t)DQ * DQ * 2);       // 0.5 MB
  __bf16* cf_bf = (__bf16*)alloc((size_t)BQ * SQ * DQ * 2);  // 4 MB
  float*  cs    = (float*) alloc((size_t)BQ * TQ * SQ * 4);  // 2 MB

  auto cvt = [&](const float* srcp, __bf16* dstp, int n) {
    f32_to_bf16_kernel<<<(n + 255) / 256, 256, 0, stream>>>(srcp, dstp, n);
  };
  cvt(s,    s_bf,  BQ * TQ * DQ);
  cvt(h,    h_bf,  BQ * SQ * DQ);
  cvt(Wo_w, Wo_bf, VQ * DQ);
  cvt(Wc_w, Wc_bf, DQ * DQ);

  copy_feat_gemm<<<dim3(DQ / 128, (BQ * SQ) / 64), 256, 0, stream>>>(h_bf, Wc_bf, Wc_b, cf_bf);
  gen_scores_gemm<<<dim3(VQ / 128, (BQ * TQ) / 64), 256, 0, stream>>>(s_bf, Wo_bf, Wo_b, out);
  copy_scores_gemm<<<dim3(SQ / 128, TQ / 64, BQ), 256, 0, stream>>>(s_bf, cf_bf, cs);

  hipFuncSetAttribute(reinterpret_cast<const void*>(softmax_scatter_kernel),
                      hipFuncAttributeMaxDynamicSharedMemorySize, VQ * 4);
  softmax_scatter_kernel<<<BQ * TQ, 512, VQ * 4, stream>>>(cs, seq, out);
}
